// Model_71021579207151
// MI455X (gfx1250) — matrix operand order repaired and checked by reading, not compiled, not run on hardware
//
#include <hip/hip_runtime.h>
#include <hip/hip_bf16.h>

// ---------------------------------------------------------------------------
// Types / helpers
// ---------------------------------------------------------------------------
typedef __attribute__((ext_vector_type(16))) __bf16          bf16x16;
typedef __attribute__((ext_vector_type(8)))  float           f32x8;
typedef __attribute__((ext_vector_type(16))) unsigned short  u16x16;

#define DEV __device__ __forceinline__

DEV unsigned short f2bf(float f) {                 // fp32 -> bf16 (RNE)
  unsigned int u = __builtin_bit_cast(unsigned int, f);
  u += 0x7FFFu + ((u >> 16) & 1u);
  return (unsigned short)(u >> 16);
}

DEV float gelu_tanh(float x) {                     // jax.nn.gelu (approximate)
  float x3 = x * x * x;
  return 0.5f * x * (1.0f + tanhf(0.7978845608028654f * (x + 0.044715f * x3)));
}

DEV f32x8 wmma_bf16(u16x16 a, u16x16 b, f32x8 c) {
  return __builtin_amdgcn_wmma_f32_16x16x32_bf16(
      false, __builtin_bit_cast(bf16x16, a),
      false, __builtin_bit_cast(bf16x16, b),
      (short)0, c, false, false);
}

// 16-bit A-matrix 16x32 wave32 layout (contiguous k order, matching B fills): element e of lane-half ->
// K index within the 32-wide slab.
DEV int a_kpat(int e, int lhalf) {
  return lhalf * 16 + e;
}

// Async global -> LDS copy of 16 bytes (GLOBAL_LOAD_ASYNC_TO_LDS_B128,
// tracked by ASYNCcnt).  ldsoff = 32-bit LDS byte offset (generic shared
// pointer truncated: LDS aperture keeps the offset in addr[31:0]).
DEV void async_copy_b128(const void* gptr, void* sptr) {
  unsigned lds = (unsigned)(unsigned long long)sptr;
  unsigned long long g = (unsigned long long)gptr;
  asm volatile("global_load_async_to_lds_b128 %0, %1, off"
               :: "v"(lds), "v"(g) : "memory");
}
DEV void async_wait0() {
  asm volatile("s_wait_asynccnt 0x0" ::: "memory");
}

// ---------------------------------------------------------------------------
// fp32 -> bf16 weight conversion
// ---------------------------------------------------------------------------
__global__ void cvt_k(const float* __restrict__ in, unsigned short* __restrict__ out, int n) {
  for (int i = blockIdx.x * blockDim.x + threadIdx.x; i < n; i += gridDim.x * blockDim.x)
    out[i] = f2bf(in[i]);
}

// ---------------------------------------------------------------------------
// BigBird block masks: masks[layer*16 + qblock] = 16-bit key-block bitmask
// ---------------------------------------------------------------------------
__global__ void build_masks_k(unsigned int* __restrict__ masks) {
  int t = blockIdx.x * blockDim.x + threadIdx.x;
  if (t >= 6 * 16) return;
  int layer = t >> 4, i = t & 15;
  unsigned int m;
  if (i == 0 || i == 15) {
    m = 0xFFFFu;                                   // global rows attend everywhere
  } else {
    m = 0x8001u;                                   // global cols 0 and 15
    m |= 7u << (i - 1);                            // sliding window i-1..i+1
    unsigned int st = (unsigned)(layer * 1103515245 + i * 40503 + 12345);
    int cnt = 0;
    while (cnt < 3) {                              // 3 random middle blocks
      st = st * 1664525u + 1013904223u;
      int j = 1 + (int)((st >> 16) % 14u);         // j in [1,14]
      int d = j - i; if (d < 0) d = -d;
      if (d > 1 && !((m >> j) & 1u)) { m |= 1u << j; ++cnt; }
    }
  }
  masks[t] = m;
}

// ---------------------------------------------------------------------------
// Embedding gather + positional add -> fp32 residual stream x[B,S,384]
// ---------------------------------------------------------------------------
__global__ void embed_k(const int* __restrict__ tc, const int* __restrict__ tw,
                        const float* __restrict__ ce, const float* __restrict__ we,
                        const float* __restrict__ pe, float* __restrict__ x, int n) {
  for (int idx = blockIdx.x * blockDim.x + threadIdx.x; idx < n; idx += gridDim.x * blockDim.x) {
    int d = idx % 384;
    int bs = idx / 384;                            // b*1024 + s
    int s = bs & 1023;
    float v = (d < 192) ? ce[(size_t)tc[bs] * 192 + d]
                        : we[(size_t)tw[bs] * 192 + (d - 192)];
    x[idx] = v + pe[s * 384 + d];
  }
}

// ---------------------------------------------------------------------------
// LayerNorm(D=384), one wave32 per token, output bf16
// ---------------------------------------------------------------------------
__global__ __launch_bounds__(256) void ln_k(const float* __restrict__ x,
                                            const float* __restrict__ sc,
                                            const float* __restrict__ bi,
                                            unsigned short* __restrict__ out, int ntok) {
  int wid = threadIdx.x >> 5, lane = threadIdx.x & 31;
  int tok = blockIdx.x * 8 + wid;
  if (tok >= ntok) return;
  const float* xp = x + (size_t)tok * 384;
  float v[12];
  float sum = 0.f;
  for (int j = 0; j < 12; ++j) { v[j] = xp[lane + 32 * j]; sum += v[j]; }
  for (int off = 16; off >= 1; off >>= 1) sum += __shfl_xor(sum, off, 32);
  float mu = sum * (1.0f / 384.0f);
  float sq = 0.f;
  for (int j = 0; j < 12; ++j) { float d = v[j] - mu; sq += d * d; }
  for (int off = 16; off >= 1; off >>= 1) sq += __shfl_xor(sq, off, 32);
  float r = rsqrtf(sq * (1.0f / 384.0f) + 1e-6f);
  unsigned short* op = out + (size_t)tok * 384;
  for (int j = 0; j < 12; ++j) {
    int d = lane + 32 * j;
    op[d] = f2bf((v[j] - mu) * r * sc[d] + bi[d]);
  }
}

// ---------------------------------------------------------------------------
// Generic bf16 WMMA GEMM: C[M,N] = A[M,K] * B[K,N] (+bias)(gelu)(+resid)
// 256 threads = 8 waves (4x2); block tile 128x64, wave tile 32x32, K step 32.
// Tiles staged via GLOBAL_LOAD_ASYNC_TO_LDS_B128 (ASYNCcnt).
// ---------------------------------------------------------------------------
template <bool BIAS, bool GELU, bool RESID, bool OUTBF>
__global__ __launch_bounds__(256) void gemm_bf16_k(
    const unsigned short* __restrict__ A, const unsigned short* __restrict__ Bm,
    const float* __restrict__ bias, const float* __restrict__ resid,
    float* __restrict__ Cf, unsigned short* __restrict__ Cb,
    int M, int N, int K) {
  __shared__ unsigned short sA[128 * 32];
  __shared__ unsigned short sB[32 * 64];
  const int tid = threadIdx.x;
  const int wid = tid >> 5, lane = tid & 31, lhalf = lane >> 4, lm = lane & 15;
  const int m0 = blockIdx.y * 128, n0 = blockIdx.x * 64;
  const int wm = (wid & 3) * 32, wn = (wid >> 2) * 32;
  const bool nvec = ((N & 7) == 0);

  f32x8 zero = {0.f, 0.f, 0.f, 0.f, 0.f, 0.f, 0.f, 0.f};
  f32x8 acc[2][2] = {{zero, zero}, {zero, zero}};

  for (int k0 = 0; k0 < K; k0 += 32) {
    // A tile 128x32: two async b128 copies per thread
    for (int i = 0; i < 2; ++i) {
      int idx = tid + i * 256;
      int row = idx >> 2, cg = (idx & 3) << 3;
      async_copy_b128(&A[(size_t)(m0 + row) * K + k0 + cg], &sA[row * 32 + cg]);
    }
    // B tile 32x64: async b128 when in range / aligned, scalar fallback else
    {
      int row = tid >> 3, cg = (tid & 7) << 3;
      if (nvec && (n0 + cg + 8 <= N)) {
        async_copy_b128(&Bm[(size_t)(k0 + row) * N + n0 + cg], &sB[row * 64 + cg]);
      } else {
        for (int j = 0; j < 8; ++j) {
          int col = n0 + cg + j;
          sB[row * 64 + cg + j] =
              (col < N) ? Bm[(size_t)(k0 + row) * N + col] : (unsigned short)0;
        }
      }
    }
    async_wait0();
    __syncthreads();
    if (k0 + 32 < K)                               // global_prefetch_b8 next A slab
      __builtin_prefetch(&A[(size_t)(m0 + (tid >> 1)) * K + k0 + 32], 0, 0);

    u16x16 af[2], b0f, b1f;
    for (int e = 0; e < 16; ++e) {
      int kp = a_kpat(e, lhalf);
      af[0][e] = sA[(wm + lm) * 32 + kp];
      af[1][e] = sA[(wm + 16 + lm) * 32 + kp];
      int kk = lhalf * 16 + e;
      b0f[e] = sB[kk * 64 + wn + lm];
      b1f[e] = sB[kk * 64 + wn + 16 + lm];
    }
    acc[0][0] = wmma_bf16(af[0], b0f, acc[0][0]);
    acc[0][1] = wmma_bf16(af[0], b1f, acc[0][1]);
    acc[1][0] = wmma_bf16(af[1], b0f, acc[1][0]);
    acc[1][1] = wmma_bf16(af[1], b1f, acc[1][1]);
    __syncthreads();
  }

  for (int tr = 0; tr < 2; ++tr)                   // epilogue
    for (int tc = 0; tc < 2; ++tc) {
      int col = n0 + wn + tc * 16 + lm;
      if (col >= N) continue;
      for (int e = 0; e < 8; ++e) {
        int row = m0 + wm + tr * 16 + lhalf * 8 + e;   // C layout: M = e + 8*lhalf
        float v = acc[tr][tc][e];
        if (BIAS)  v += bias[col];
        if (GELU)  v = gelu_tanh(v);
        if (RESID) v += resid[(size_t)row * N + col];
        if (OUTBF) Cb[(size_t)row * N + col] = f2bf(v);
        else       Cf[(size_t)row * N + col] = v;
      }
    }
}

// ---------------------------------------------------------------------------
// Block-sparse flash attention. One workgroup per (qblock, head, batch).
// 128 threads = 4 waves, 16 q-rows/wave, HD=64. Q/K/V layout [B,S,H*64] bf16.
// K/V tiles staged via async-to-LDS copies.
// ---------------------------------------------------------------------------
__global__ __launch_bounds__(128) void attn_k(
    const unsigned short* __restrict__ Q, const unsigned short* __restrict__ Kt,
    const unsigned short* __restrict__ V, const int* __restrict__ tchar,
    const unsigned int* __restrict__ masks, unsigned short* __restrict__ O, int layer) {
  __shared__ unsigned short sK[64 * 64];
  __shared__ unsigned short sV[64 * 64];
  __shared__ unsigned short sP[4][16 * 64];
  const int qb = blockIdx.x, h = blockIdx.y, b = blockIdx.z;
  const int tid = threadIdx.x, w = tid >> 5, lane = tid & 31;
  const int lhalf = lane >> 4, lm = lane & 15;
  const unsigned int mask = masks[layer * 16 + qb];

  u16x16 qf[2];                                    // Q A-fragments (d = 0..63)
  {
    size_t qrow = (size_t)(b * 1024 + qb * 64 + w * 16 + lm) * 512 + h * 64;
    for (int ds = 0; ds < 2; ++ds)
      for (int e = 0; e < 16; ++e)
        qf[ds][e] = Q[qrow + ds * 32 + a_kpat(e, lhalf)];
  }

  f32x8 zero = {0.f, 0.f, 0.f, 0.f, 0.f, 0.f, 0.f, 0.f};
  float mrow[8], lrow[8];
  f32x8 acc[4];
  for (int e = 0; e < 8; ++e) { mrow[e] = -1e30f; lrow[e] = 0.f; }
  for (int n = 0; n < 4; ++n) acc[n] = zero;

  for (int kb = 0; kb < 16; ++kb) {
    if (!((mask >> kb) & 1u)) continue;            // uniform across workgroup
    __syncthreads();
    for (int i = 0; i < 4; ++i) {                  // async K/V tile fills
      int idx = tid + i * 128;
      int row = idx >> 3, cg = (idx & 7) << 3;
      size_t g = (size_t)(b * 1024 + kb * 64 + row) * 512 + h * 64 + cg;
      async_copy_b128(&Kt[g], &sK[row * 64 + cg]);
      async_copy_b128(&V[g],  &sV[row * 64 + cg]);
    }
    async_wait0();
    __syncthreads();

    f32x8 s[4];                                    // S = Q K^T / 8, key-pad mask
    for (int t = 0; t < 4; ++t) {
      f32x8 c = zero;
      for (int ds = 0; ds < 2; ++ds) {
        u16x16 bf;
        for (int e = 0; e < 16; ++e)               // B[K=d, N=kcol] = K[kcol, d]
          bf[e] = sK[(t * 16 + lm) * 64 + ds * 32 + lhalf * 16 + e];
        c = wmma_bf16(qf[ds], bf, c);
      }
      bool valid = tchar[b * 1024 + kb * 64 + t * 16 + lm] > 0;
      for (int e = 0; e < 8; ++e)
        s[t][e] = valid ? c[e] * 0.125f : -1e9f;
    }

    for (int e = 0; e < 8; ++e) {                  // online softmax: new row max
      float r = fmaxf(fmaxf(s[0][e], s[1][e]), fmaxf(s[2][e], s[3][e]));
      for (int off = 8; off >= 1; off >>= 1)
        r = fmaxf(r, __shfl_xor(r, off, 32));
      float nm = fmaxf(mrow[e], r);
      float al = __expf(mrow[e] - nm);
      lrow[e] *= al;
      for (int n = 0; n < 4; ++n) acc[n][e] *= al;
      mrow[e] = nm;
    }
    for (int t = 0; t < 4; ++t)
      for (int e = 0; e < 8; ++e)
        s[t][e] = __expf(s[t][e] - mrow[e]);
    for (int e = 0; e < 8; ++e) {
      float r = s[0][e] + s[1][e] + s[2][e] + s[3][e];
      for (int off = 8; off >= 1; off >>= 1)
        r += __shfl_xor(r, off, 32);
      lrow[e] += r;
    }

    // C-layout P -> LDS -> A-layout fragments (same-wave LDS is in-order)
    for (int t = 0; t < 4; ++t)
      for (int e = 0; e < 8; ++e)
        sP[w][(e + lhalf * 8) * 64 + t * 16 + lm] = f2bf(s[t][e]);
    u16x16 pf[2];
    for (int ks = 0; ks < 2; ++ks)
      for (int e = 0; e < 16; ++e)
        pf[ks][e] = sP[w][lm * 64 + ks * 32 + a_kpat(e, lhalf)];

    for (int n = 0; n < 4; ++n)                    // O += P @ V
      for (int ks = 0; ks < 2; ++ks) {
        u16x16 vf;
        for (int e = 0; e < 16; ++e)
          vf[e] = sV[(ks * 32 + lhalf * 16 + e) * 64 + n * 16 + lm];
        acc[n] = wmma_bf16(pf[ks], vf, acc[n]);
      }
  }

  for (int e = 0; e < 8; ++e) {                    // normalize + store bf16
    float inv = 1.0f / fmaxf(lrow[e], 1e-20f);
    size_t row = (size_t)(b * 1024 + qb * 64 + w * 16 + lhalf * 8 + e) * 512 + h * 64;
    for (int n = 0; n < 4; ++n)
      O[row + n * 16 + lm] = f2bf(acc[n][e] * inv);
  }
}

// ---------------------------------------------------------------------------
// Host orchestration
// ---------------------------------------------------------------------------
extern "C" void kernel_launch(void* const* d_in, const int* in_sizes, int n_in,
                              void* d_out, int out_size, void* d_ws, size_t ws_size,
                              hipStream_t stream) {
  (void)in_sizes; (void)n_in; (void)out_size; (void)ws_size;
  constexpr int S = 1024, D = 384, LL = 6, MLPD = 2048, EMBD = 512;
  constexpr int M = 4 * S;                         // 4096 tokens

  // jax pytree flatten (dict keys sorted alphabetically)
  const int*   tchar    = (const int*)d_in[0];
  const int*   tword    = (const int*)d_in[1];
  const float* char_emb = (const float*)d_in[2];
  // heads: date, mask, nsp, sub; each b1,b2,w1,w2
  const float* h_b1[4] = {(const float*)d_in[3],  (const float*)d_in[7],
                          (const float*)d_in[11], (const float*)d_in[15]};
  const float* h_b2[4] = {(const float*)d_in[4],  (const float*)d_in[8],
                          (const float*)d_in[12], (const float*)d_in[16]};
  const float* h_w1[4] = {(const float*)d_in[5],  (const float*)d_in[9],
                          (const float*)d_in[13], (const float*)d_in[17]};
  const float* h_w2[4] = {(const float*)d_in[6],  (const float*)d_in[10],
                          (const float*)d_in[14], (const float*)d_in[18]};
  const int headN[4] = {160, 192, 2, 85};          // date, mask, nsp, sub
  const float* l_b1   = (const float*)d_in[19];
  const float* l_b2   = (const float*)d_in[20];
  const float* l_ln1b = (const float*)d_in[21];
  const float* l_ln1s = (const float*)d_in[22];
  const float* l_ln2b = (const float*)d_in[23];
  const float* l_ln2s = (const float*)d_in[24];
  const float* l_w1   = (const float*)d_in[25];
  const float* l_w2   = (const float*)d_in[26];
  const float* l_wk   = (const float*)d_in[27];
  const float* l_wo   = (const float*)d_in[28];
  const float* l_wq   = (const float*)d_in[29];
  const float* l_wv   = (const float*)d_in[30];
  const float* lnf_b  = (const float*)d_in[31];
  const float* lnf_s  = (const float*)d_in[32];
  const float* pos    = (const float*)d_in[33];
  const float* word_emb = (const float*)d_in[34];
  float* out = (float*)d_out;

  size_t off = 0;
  auto alloc = [&](size_t bytes) -> char* {
    char* p = (char*)d_ws + off;
    off += (bytes + 255) & ~(size_t)255;
    return p;
  };
  unsigned short* wq_b  = (unsigned short*)alloc((size_t)LL * D * EMBD * 2);
  unsigned short* wk_b  = (unsigned short*)alloc((size_t)LL * D * EMBD * 2);
  unsigned short* wv_b  = (unsigned short*)alloc((size_t)LL * D * EMBD * 2);
  unsigned short* wo_b  = (unsigned short*)alloc((size_t)LL * EMBD * D * 2);
  unsigned short* w1_b  = (unsigned short*)alloc((size_t)LL * D * MLPD * 2);
  unsigned short* w2_b  = (unsigned short*)alloc((size_t)LL * MLPD * D * 2);
  unsigned short* hw1_b = (unsigned short*)alloc((size_t)4 * D * EMBD * 2);
  unsigned short* hw2_b[4];
  for (int hi = 0; hi < 4; ++hi)
    hw2_b[hi] = (unsigned short*)alloc((size_t)EMBD * headN[hi] * 2);
  float*          x     = (float*)alloc((size_t)M * D * 4);
  unsigned short* hln   = (unsigned short*)alloc((size_t)M * D * 2);
  unsigned short* qb    = (unsigned short*)alloc((size_t)M * EMBD * 2);
  unsigned short* kb    = (unsigned short*)alloc((size_t)M * EMBD * 2);
  unsigned short* vb    = (unsigned short*)alloc((size_t)M * EMBD * 2);
  unsigned short* attnb = (unsigned short*)alloc((size_t)M * EMBD * 2);
  unsigned short* mlp1  = (unsigned short*)alloc((size_t)M * MLPD * 2);
  unsigned short* headt = (unsigned short*)alloc((size_t)M * EMBD * 2);
  unsigned int*   masks = (unsigned int*)alloc(LL * 16 * 4);

  auto CVT = [&](const float* src, unsigned short* dst, int n) {
    int g = (n + 255) / 256; if (g > 2048) g = 2048;
    cvt_k<<<g, 256, 0, stream>>>(src, dst, n);
  };
  CVT(l_wq, wq_b, LL * D * EMBD);
  CVT(l_wk, wk_b, LL * D * EMBD);
  CVT(l_wv, wv_b, LL * D * EMBD);
  CVT(l_wo, wo_b, LL * EMBD * D);                  // (H,HD,D) flat = (512,384)
  CVT(l_w1, w1_b, LL * D * MLPD);
  CVT(l_w2, w2_b, LL * MLPD * D);
  for (int hi = 0; hi < 4; ++hi) {
    CVT(h_w1[hi], hw1_b + (size_t)hi * D * EMBD, D * EMBD);
    CVT(h_w2[hi], hw2_b[hi], EMBD * headN[hi]);
  }
  build_masks_k<<<1, 128, 0, stream>>>(masks);
  embed_k<<<1024, 256, 0, stream>>>(tchar, tword, char_emb, word_emb, pos, x, M * D);

  for (int l = 0; l < LL; ++l) {
    ln_k<<<M / 8, 256, 0, stream>>>(x, l_ln1s + l * D, l_ln1b + l * D, hln, M);
    gemm_bf16_k<false, false, false, true><<<dim3(EMBD / 64, M / 128), 256, 0, stream>>>(
        hln, wq_b + (size_t)l * D * EMBD, nullptr, nullptr, nullptr, qb, M, EMBD, D);
    gemm_bf16_k<false, false, false, true><<<dim3(EMBD / 64, M / 128), 256, 0, stream>>>(
        hln, wk_b + (size_t)l * D * EMBD, nullptr, nullptr, nullptr, kb, M, EMBD, D);
    gemm_bf16_k<false, false, false, true><<<dim3(EMBD / 64, M / 128), 256, 0, stream>>>(
        hln, wv_b + (size_t)l * D * EMBD, nullptr, nullptr, nullptr, vb, M, EMBD, D);
    attn_k<<<dim3(16, 8, 4), 128, 0, stream>>>(qb, kb, vb, tchar, masks, attnb, l);
    gemm_bf16_k<false, false, true, false><<<dim3(D / 64, M / 128), 256, 0, stream>>>(
        attnb, wo_b + (size_t)l * EMBD * D, nullptr, x, x, nullptr, M, D, EMBD);
    ln_k<<<M / 8, 256, 0, stream>>>(x, l_ln2s + l * D, l_ln2b + l * D, hln, M);
    gemm_bf16_k<true, true, false, true><<<dim3(MLPD / 64, M / 128), 256, 0, stream>>>(
        hln, w1_b + (size_t)l * D * MLPD, l_b1 + l * MLPD, nullptr, nullptr, mlp1, M, MLPD, D);
    gemm_bf16_k<true, false, true, false><<<dim3(D / 64, M / 128), 256, 0, stream>>>(
        mlp1, w2_b + (size_t)l * MLPD * D, l_b2 + l * D, x, x, nullptr, M, D, MLPD);
  }

  ln_k<<<M / 8, 256, 0, stream>>>(x, lnf_s, lnf_b, hln, M);

  const int order[4] = {0, 3, 1, 2};               // output order: date, sub, mask, nsp
  float* outp = out;
  for (int oi = 0; oi < 4; ++oi) {
    int hi = order[oi], N = headN[hi];
    gemm_bf16_k<true, true, false, true><<<dim3(EMBD / 64, M / 128), 256, 0, stream>>>(
        hln, hw1_b + (size_t)hi * D * EMBD, h_b1[hi], nullptr, nullptr, headt, M, EMBD, D);
    gemm_bf16_k<true, false, false, false><<<dim3((N + 63) / 64, M / 128), 256, 0, stream>>>(
        headt, hw2_b[hi], h_b2[hi], nullptr, outp, nullptr, M, N, EMBD);
    outp += (size_t)M * N;
  }
}
